// DGLJTNNDecoder_65489661329578
// MI455X (gfx1250) — matrix operand order repaired and checked by reading, not compiled, not run on hardware
//
#include <hip/hip_runtime.h>
#include <stdint.h>

// Problem constants (match reference)
#define Bc 2048
#define Dc 24
#define Hc 256
#define Lc 64
#define Vc 780
#define DBr (Dc * Bc)              // 49152 q rows
#define PRr ((2 * Dc - 1) * Bc)    // 96256 p rows
#define BHe ((long)Bc * Hc)        // 524288 elems per [B,H] slab

// WMMA GEMM tiling
#define BM 128
#define BN 64
#define BK 32
#define LDP 20                     // LDS row stride in dwords: conflict-free + 16B-aligned

typedef __attribute__((ext_vector_type(16))) __bf16 v16bf;
typedef __attribute__((ext_vector_type(8)))  float  v8f;

union Frag16 { uint32_t u[8]; v16bf v; };

__device__ __forceinline__ uint16_t f2bf(float f) {
  uint32_t x = __float_as_uint(f);
  x += 0x7FFFu + ((x >> 16) & 1u);        // round-to-nearest-even
  return (uint16_t)(x >> 16);
}
__device__ __forceinline__ float bf2f(uint16_t h) {
  return __uint_as_float(((uint32_t)h) << 16);
}
__device__ __forceinline__ float sigmoidf_(float x) {
  return 1.0f / (1.0f + expf(-x));
}

// ---------------------------------------------------------------------------
// Generic bf16 WMMA GEMM:  C[M,N] = A1[M,0:K1]@W1^T + A2[M,0:K2]@W2^T + bias
// A row-major (u16 bf16), W row-major [N, K] (u16 bf16).
// Cf != null -> f32 output; else bf16 output to Cb. act==1 -> ReLU.
// LAUNCH INVARIANTS (all call sites satisfy these):
//   M % BM == 0; K1 % BK == 0; K2 % BK == 0;
//   A/W/C base pointers 256B-aligned; lda/ldw/ldc even (16B-aligned rows).
// Only N may be ragged (V = 780).
// ---------------------------------------------------------------------------
__global__ __launch_bounds__(256)
void gemm_bf16_wmma(const uint16_t* __restrict__ A1, int lda1, int K1,
                    const uint16_t* __restrict__ A2, int lda2, int K2,
                    const uint16_t* __restrict__ W1, int ldw1,
                    const uint16_t* __restrict__ W2, int ldw2,
                    const float* __restrict__ bias,
                    float* __restrict__ Cf, uint16_t* __restrict__ Cb,
                    int ldc, int M, int N, int act)
{
  __shared__ __align__(16) uint32_t As[BM][LDP];   // 128 x 16 dwords used
  __shared__ __align__(16) uint32_t Bs[BN][LDP];   //  64 x 16 dwords used

  const int tid  = threadIdx.x;
  const int lane = tid & 31;
  const int wv   = tid >> 5;               // wave 0..7
  const int half = lane >> 4;
  const int lrow = lane & 15;
  const int m_off = (wv >> 1) * 32;        // 4 waves over M
  const int n_off = (wv & 1) * 32;         // 2 waves over N
  const int row0 = blockIdx.y * BM;
  const int col0 = blockIdx.x * BN;
  const int Ktot = K1 + K2;

  v8f acc[2][2] = {};

  for (int kc = 0; kc < Ktot; kc += BK) {
    // Uniform (scalar) K-source select: chunks never straddle the K1 boundary.
    const uint16_t* Ap; int ldaA; int ka0;
    const uint16_t* Wp; int ldwW; int kw0;
    if (kc < K1) { Ap = A1; ldaA = lda1; ka0 = kc;      Wp = W1; ldwW = ldw1; kw0 = kc; }
    else         { Ap = A2; ldaA = lda2; ka0 = kc - K1; Wp = W2; ldwW = ldw2; kw0 = kc - K1; }

    // ---- stage A tile: 128 rows x 16 dwords = 512 uint4, 2 per thread ----
#pragma unroll
    for (int it = 0; it < 2; ++it) {
      int idx = tid + it * 256;            // 0..511
      int r   = idx >> 2;
      int c4  = (idx & 3) * 4;             // dword column: 0,4,8,12 (16B aligned)
      uint4 v = *(const uint4*)(Ap + (size_t)(row0 + r) * ldaA + ka0 + c4 * 2);
      *(uint4*)&As[r][c4] = v;
    }
    // ---- stage W tile: 64 rows x 16 dwords = 256 uint4, 1 per thread ----
    {
      int n  = tid >> 2;
      int c4 = (tid & 3) * 4;
      uint4 v = make_uint4(0u, 0u, 0u, 0u);
      if (col0 + n < N)
        v = *(const uint4*)(Wp + (size_t)(col0 + n) * ldwW + kw0 + c4 * 2);
      *(uint4*)&Bs[n][c4] = v;
    }
    // light prefetch of next K-chunk of A (global_prefetch path)
    if (kc + BK < Ktot) {
      int nk = kc + BK;
      const uint16_t* Pp; int ldp; int kp;
      if (nk < K1) { Pp = A1; ldp = lda1; kp = nk; }
      else         { Pp = A2; ldp = lda2; kp = nk - K1; }
      __builtin_prefetch(Pp + (size_t)(row0 + (tid >> 1)) * ldp + kp + (tid & 1) * 16, 0, 1);
    }
    __syncthreads();

    // ---- build fragments per CDNA5 WMMA VGPR layouts ----
    Frag16 af[2], bfr[2];
#pragma unroll
    for (int s = 0; s < 2; ++s) {
      int am = m_off + s * 16 + lrow;      // A: M = lane&15 for both halves
#pragma unroll
      for (int v = 0; v < 8; ++v) {
        // A 16x32 bf16: VGPR0-3: K pairs {0..7}+8*half ; VGPR4-7: {16..23}+8*half
        int kd = (v < 4 ? v : v + 4) + 4 * half;
        af[s].u[v] = As[am][kd];
      }
      int bn = n_off + s * 16 + lrow;      // B: N = lane&15
#pragma unroll
      for (int v = 0; v < 8; ++v)
        bfr[s].u[v] = Bs[bn][(v < 4 ? v : v + 4) + 4 * half];  // same K order as A: {0..7,16..23}+8*half
    }
#pragma unroll
    for (int si = 0; si < 2; ++si)
#pragma unroll
      for (int sj = 0; sj < 2; ++sj)
        acc[si][sj] = __builtin_amdgcn_wmma_f32_16x16x32_bf16(
            false, af[si].v, false, bfr[sj].v, (short)0, acc[si][sj], false, false);
    __syncthreads();
  }

  // ---- epilogue: C layout lane0-15 M=r, lane16-31 M=8+r, N=lane&15 ----
#pragma unroll
  for (int si = 0; si < 2; ++si) {
#pragma unroll
    for (int sj = 0; sj < 2; ++sj) {
      int nc = col0 + n_off + sj * 16 + lrow;
      if (nc >= N) continue;
      float bv = bias ? bias[nc] : 0.0f;
      int mbase = row0 + m_off + si * 16 + half * 8;   // M % BM == 0: no row guard
#pragma unroll
      for (int r = 0; r < 8; ++r) {
        float v = acc[si][sj][r] + bv;
        if (act == 1) v = fmaxf(v, 0.0f);
        if (Cf) Cf[(size_t)(mbase + r) * ldc + nc] = v;
        else    Cb[(size_t)(mbase + r) * ldc + nc] = f2bf(v);
      }
    }
  }
}

// ---------------------------------------------------------------------------
// Small utility / elementwise kernels
// ---------------------------------------------------------------------------
__global__ void zero_u32_k(uint32_t* p, long n) {
  long i = (long)blockIdx.x * blockDim.x + threadIdx.x;
  if (i < n) p[i] = 0u;
}

__global__ void f2bf_copy_k(const float* __restrict__ s, uint16_t* __restrict__ d, long n) {
  long i = (long)blockIdx.x * blockDim.x + threadIdx.x;
  if (i < n) d[i] = f2bf(s[i]);
}

// x_bf16[t,b,:] = bf16(emb[wid[b,t], :])
__global__ void gather_embed_k(const int* __restrict__ wid, const float* __restrict__ emb,
                               uint16_t* __restrict__ xb, long n) {
  long i = (long)blockIdx.x * blockDim.x + threadIdx.x;
  if (i >= n) return;
  int c = (int)(i % Hc);
  long tb = i / Hc;
  int b = (int)(tb % Bc);
  int t = (int)(tb / Bc);
  int w = wid[(size_t)b * Dc + t];
  xb[i] = f2bf(emb[(size_t)w * Hc + c]);
}

// m = (1-sigmoid(z))*m_prev + sigmoid(z)*tanh(h); write f32 carry, bf16, and m_out
__global__ void gru_update_m_k(const float* __restrict__ zp, const float* __restrict__ hp,
                               float* __restrict__ m_state, uint16_t* __restrict__ m_b16,
                               float* __restrict__ m_out, long n) {
  long i = (long)blockIdx.x * blockDim.x + threadIdx.x;
  if (i >= n) return;
  float z  = sigmoidf_(zp[i]);
  float mt = tanhf(hp[i]);
  float m  = (1.0f - z) * m_state[i] + z * mt;
  m_state[i] = m;
  m_b16[i]   = f2bf(m);
  m_out[i]   = m;
}

// rm = sigmoid(r_pre) * m  (bf16 carry for next step's Wh input)
__global__ void gru_update_rm_k(const float* __restrict__ rp, const float* __restrict__ m_state,
                                uint16_t* __restrict__ rm_b16, long n) {
  long i = (long)blockIdx.x * blockDim.x + threadIdx.x;
  if (i >= n) return;
  rm_b16[i] = f2bf(sigmoidf_(rp[i]) * m_state[i]);
}

// q_in[t*B+b, :] = [ (t==0 ? 0 : m_f[t-1,b,:]), tree_vec[b,:] ]   width H+L=320
__global__ void build_q_in_k(const float* __restrict__ mf, const float* __restrict__ tree,
                             uint16_t* __restrict__ qin) {
  long idx = (long)blockIdx.x * blockDim.x + threadIdx.x;
  const long total = (long)DBr * (Hc + Lc);
  if (idx >= total) return;
  int  c   = (int)(idx % (Hc + Lc));
  long row = idx / (Hc + Lc);
  int  t = (int)(row / Bc), b = (int)(row % Bc);
  float v;
  if (c < Hc) v = (t == 0) ? 0.0f : mf[((size_t)(t - 1) * Bc + b) * Hc + c];
  else        v = tree[(size_t)b * Lc + (c - Hc)];
  qin[idx] = f2bf(v);
}

// p_in rows: [root(B)] [down (D-1)*B] [up (D-1)*B], width 2H+L=576
__global__ void build_p_in_k(const uint16_t* __restrict__ xb, const float* __restrict__ mf,
                             const float* __restrict__ mr, const float* __restrict__ tree,
                             uint16_t* __restrict__ pin) {
  long idx = (long)blockIdx.x * blockDim.x + threadIdx.x;
  const int W = 2 * Hc + Lc;
  const long total = (long)PRr * W;
  if (idx >= total) return;
  int  c   = (int)(idx % W);
  long row = idx / W;
  int  b = (int)(row % Bc);
  uint16_t v;
  if (c >= 2 * Hc) {
    v = f2bf(tree[(size_t)b * Lc + (c - 2 * Hc)]);
  } else if (row < Bc) {                      // root: [x_t0, zeros, tv]
    v = (c < Hc) ? xb[(size_t)b * Hc + c] : (uint16_t)0;
  } else if (row < (long)Dc * Bc) {           // down: [x_{t+1}, m_f[t], tv]
    int t = (int)(row / Bc) - 1;
    if (c < Hc) v = xb[((size_t)(t + 1) * Bc + b) * Hc + c];
    else        v = f2bf(mf[((size_t)t * Bc + b) * Hc + (c - Hc)]);
  } else {                                    // up: [x_{D-2-i}, h_up[i], tv]
    int i = (int)((row - (long)Dc * Bc) / Bc);
    if (c < Hc) v = xb[((size_t)(Dc - 2 - i) * Bc + b) * Hc + c];
    else {
      float hv = mr[((size_t)i * Bc + b) * Hc + (c - Hc)];
      if (i != Dc - 2) hv += mf[((size_t)(Dc - 3 - i) * Bc + b) * Hc + (c - Hc)];
      v = f2bf(hv);
    }
  }
  pin[idx] = v;
}

// Per-row logsumexp + selected logit + argmax accuracy over V=780 logits
__global__ __launch_bounds__(256)
void q_reduce_k(const float* __restrict__ q, const int* __restrict__ wid,
                float* __restrict__ lossA, float* __restrict__ corrA) {
  __shared__ float smax[256];
  __shared__ int   sidx[256];
  __shared__ float ssum[256];
  int row = blockIdx.x;
  int tid = threadIdx.x;
  int t = row / Bc, b = row % Bc;
  int tgt = wid[(size_t)b * Dc + t];
  const float* qr = q + (size_t)row * Vc;

  float lmax = -INFINITY; int lidx = Vc;
  for (int i = tid; i < Vc; i += 256) {
    float v = qr[i];
    if (v > lmax || (v == lmax && i < lidx)) { lmax = v; lidx = i; }
  }
  smax[tid] = lmax; sidx[tid] = lidx; __syncthreads();
  for (int s = 128; s > 0; s >>= 1) {
    if (tid < s) {
      float o = smax[tid + s]; int oi = sidx[tid + s];
      if (o > smax[tid] || (o == smax[tid] && oi < sidx[tid])) { smax[tid] = o; sidx[tid] = oi; }
    }
    __syncthreads();
  }
  float gmax = smax[0]; int gidx = sidx[0];

  float ls = 0.0f;
  for (int i = tid; i < Vc; i += 256) ls += expf(qr[i] - gmax);
  ssum[tid] = ls; __syncthreads();
  for (int s = 128; s > 0; s >>= 1) { if (tid < s) ssum[tid] += ssum[tid + s]; __syncthreads(); }
  if (tid == 0) {
    float lse = gmax + logf(ssum[0]);
    lossA[row] = lse - qr[tgt];
    corrA[row] = (gidx == tgt) ? 1.0f : 0.0f;
  }
}

// p = relu-hidden dot Us_w + Us_b; BCE-with-logits term + accuracy per row
__global__ __launch_bounds__(256)
void p_dot_k(const uint16_t* __restrict__ hp, const float* __restrict__ usw,
             const float* __restrict__ usb, float* __restrict__ lossA,
             float* __restrict__ corrA) {
  __shared__ float s[256];
  int row = blockIdx.x;
  int tid = threadIdx.x;                 // Hc == blockDim == 256
  s[tid] = bf2f(hp[(size_t)row * Hc + tid]) * usw[tid];
  __syncthreads();
  for (int st = 128; st > 0; st >>= 1) { if (tid < st) s[tid] += s[tid + st]; __syncthreads(); }
  if (tid == 0) {
    float p  = s[0] + usb[0];
    float pt = (row < (Dc - 1) * Bc) ? 1.0f : 0.0f;
    lossA[row] = fmaxf(p, 0.0f) - p * pt + log1pf(expf(-fabsf(p)));
    bool pred = p > 0.0f, tgt = pt > 0.5f;
    corrA[row] = (pred == tgt) ? 1.0f : 0.0f;
  }
}

// Fixed-order deterministic final reduction -> 4 scalars
__global__ __launch_bounds__(256)
void final_reduce_k(const float* __restrict__ qlA, const float* __restrict__ qcA,
                    const float* __restrict__ plA, const float* __restrict__ pcA,
                    float* __restrict__ out) {
  __shared__ float s[256];
  int tid = threadIdx.x;
  const long nq = DBr, np = PRr;
  float r[4] = {0, 0, 0, 0};
  for (long i = tid; i < nq; i += 256) { r[0] += qlA[i]; r[2] += qcA[i]; }
  for (long i = tid; i < np; i += 256) { r[1] += plA[i]; r[3] += pcA[i]; }
  for (int j = 0; j < 4; ++j) {
    s[tid] = r[j]; __syncthreads();
    for (int st = 128; st > 0; st >>= 1) { if (tid < st) s[tid] += s[tid + st]; __syncthreads(); }
    if (tid == 0) {
      float v = s[0];
      if (j == 0)      out[0] = v / (float)Bc;      // q_loss
      else if (j == 1) out[1] = v / (float)Bc;      // p_loss
      else if (j == 2) out[2] = v / (float)nq;      // q_acc
      else             out[3] = v / (float)np;      // p_acc
    }
    __syncthreads();
  }
}

// ---------------------------------------------------------------------------
// Host orchestration. Workspace need ~350 MB (q/p intermediates time-shared).
// ---------------------------------------------------------------------------
extern "C" void kernel_launch(void* const* d_in, const int* in_sizes, int n_in,
                              void* d_out, int out_size, void* d_ws, size_t ws_size,
                              hipStream_t stream) {
  (void)in_sizes; (void)n_in; (void)out_size; (void)ws_size;
  const int*   wid  = (const int*)  d_in[0];
  const float* tree = (const float*)d_in[1];
  const float* emb  = (const float*)d_in[2];
  const float* Wz   = (const float*)d_in[3];
  const float* bz   = (const float*)d_in[4];
  const float* Wr   = (const float*)d_in[5];
  const float* Ur   = (const float*)d_in[6];
  const float* br   = (const float*)d_in[7];
  const float* Wh   = (const float*)d_in[8];
  const float* bh   = (const float*)d_in[9];
  const float* Ww   = (const float*)d_in[10];
  const float* Wb   = (const float*)d_in[11];
  const float* Uw   = (const float*)d_in[12];
  const float* Ub   = (const float*)d_in[13];
  const float* Wo   = (const float*)d_in[14];
  const float* Wob  = (const float*)d_in[15];
  const float* Usw  = (const float*)d_in[16];
  const float* Usb  = (const float*)d_in[17];
  float* out = (float*)d_out;

  char* w = (char*)d_ws;
  size_t o = 0;
  auto take = [&](size_t bytes) -> char* {
    char* p = w + o;
    o += (bytes + 255) & ~(size_t)255;
    return p;
  };

  // Persistent regions
  uint16_t* xb   = (uint16_t*)take((size_t)Dc * BHe * 2);        // [D,B,H] bf16
  uint16_t* wzb  = (uint16_t*)take((size_t)Hc * 2 * Hc * 2);
  uint16_t* whb  = (uint16_t*)take((size_t)Hc * 2 * Hc * 2);
  uint16_t* wrb  = (uint16_t*)take((size_t)Hc * Hc * 2);
  uint16_t* urb  = (uint16_t*)take((size_t)Hc * Hc * 2);
  uint16_t* wwb  = (uint16_t*)take((size_t)Hc * (Hc + Lc) * 2);
  uint16_t* uwb  = (uint16_t*)take((size_t)Hc * (2 * Hc + Lc) * 2);
  uint16_t* wob  = (uint16_t*)take((size_t)Vc * Hc * 2);
  float* mf     = (float*)take((size_t)(Dc - 1) * BHe * 4);      // m_f [D-1,B,H]
  float* mr     = (float*)take((size_t)(Dc - 1) * BHe * 4);      // m_r
  float* mstate = (float*)take(BHe * 4);
  uint16_t* mb16  = (uint16_t*)take(BHe * 2);
  uint16_t* rmb16 = (uint16_t*)take(BHe * 2);
  float* zpre = (float*)take(BHe * 4);
  float* hpre = (float*)take(BHe * 4);
  float* rpre = (float*)take(BHe * 4);
  float* qlA = (float*)take((size_t)DBr * 4);
  float* qcA = (float*)take((size_t)DBr * 4);
  float* plA = (float*)take((size_t)PRr * 4);
  float* pcA = (float*)take((size_t)PRr * 4);
  // Shared (time-multiplexed) scratch: q path then p path
  char* shared0 = take(0);
  uint16_t* qin  = (uint16_t*)shared0;                                   // DBr*320 bf16
  uint16_t* hq   = (uint16_t*)(shared0 + (((size_t)DBr * 320 * 2 + 255) & ~(size_t)255));
  float*    qmat = (float*)((char*)hq + (((size_t)DBr * Hc * 2 + 255) & ~(size_t)255));
  uint16_t* pin  = (uint16_t*)shared0;                                   // PRr*576 bf16
  uint16_t* hp   = (uint16_t*)(shared0 + (((size_t)PRr * 576 * 2 + 255) & ~(size_t)255));

  auto gemm = [&](const uint16_t* A1, int lda1, int K1,
                  const uint16_t* A2, int lda2, int K2,
                  const uint16_t* W1, int ldw1, const uint16_t* W2, int ldw2,
                  const float* bias, float* Cf, uint16_t* Cb, int ldc,
                  int M, int N, int act) {
    dim3 g((N + BN - 1) / BN, (M + BM - 1) / BM);
    gemm_bf16_wmma<<<g, 256, 0, stream>>>(A1, lda1, K1, A2, lda2, K2,
                                          W1, ldw1, W2, ldw2, bias, Cf, Cb,
                                          ldc, M, N, act);
  };
  auto ew_grid = [](long n) { return dim3((unsigned)((n + 255) / 256)); };

  // ---- 1. weights -> bf16 ----
  f2bf_copy_k<<<ew_grid((long)Hc * 2 * Hc), 256, 0, stream>>>(Wz, wzb, (long)Hc * 2 * Hc);
  f2bf_copy_k<<<ew_grid((long)Hc * 2 * Hc), 256, 0, stream>>>(Wh, whb, (long)Hc * 2 * Hc);
  f2bf_copy_k<<<ew_grid((long)Hc * Hc), 256, 0, stream>>>(Wr, wrb, (long)Hc * Hc);
  f2bf_copy_k<<<ew_grid((long)Hc * Hc), 256, 0, stream>>>(Ur, urb, (long)Hc * Hc);
  f2bf_copy_k<<<ew_grid((long)Hc * (Hc + Lc)), 256, 0, stream>>>(Ww, wwb, (long)Hc * (Hc + Lc));
  f2bf_copy_k<<<ew_grid((long)Hc * (2 * Hc + Lc)), 256, 0, stream>>>(Uw, uwb, (long)Hc * (2 * Hc + Lc));
  f2bf_copy_k<<<ew_grid((long)Vc * Hc), 256, 0, stream>>>(Wo, wob, (long)Vc * Hc);

  // ---- 2. embedding gather ----
  gather_embed_k<<<ew_grid((long)Dc * BHe), 256, 0, stream>>>(wid, emb, xb, (long)Dc * BHe);

  // ---- 3. GRU scans (forward then reverse) ----
  for (int pass = 0; pass < 2; ++pass) {
    zero_u32_k<<<ew_grid(BHe), 256, 0, stream>>>((uint32_t*)mstate, BHe);
    zero_u32_k<<<ew_grid(BHe / 2), 256, 0, stream>>>((uint32_t*)mb16, BHe / 2);
    zero_u32_k<<<ew_grid(BHe / 2), 256, 0, stream>>>((uint32_t*)rmb16, BHe / 2);
    float* mdst = (pass == 0) ? mf : mr;
    for (int t = 0; t < Dc - 1; ++t) {
      int si = (pass == 0) ? t : (Dc - 1 - t);       // src_x index
      int di = (pass == 0) ? (t + 1) : (Dc - 2 - t); // dst_x index
      const uint16_t* sx = xb + (size_t)si * BHe;
      const uint16_t* dx = xb + (size_t)di * BHe;
      // z_pre = [src_x | m] @ Wz^T + bz
      gemm(sx, Hc, Hc, mb16, Hc, Hc, wzb, 2 * Hc, wzb + Hc, 2 * Hc,
           bz, zpre, nullptr, Hc, Bc, Hc, 0);
      // h_pre = [src_x | rm] @ Wh^T + bh
      gemm(sx, Hc, Hc, rmb16, Hc, Hc, whb, 2 * Hc, whb + Hc, 2 * Hc,
           bh, hpre, nullptr, Hc, Bc, Hc, 0);
      gru_update_m_k<<<ew_grid(BHe), 256, 0, stream>>>(zpre, hpre, mstate, mb16,
                                                       mdst + (size_t)t * BHe, BHe);
      // r_pre = dst_x @ Wr^T + m @ Ur^T + br
      gemm(dx, Hc, Hc, mb16, Hc, Hc, wrb, Hc, urb, Hc,
           br, rpre, nullptr, Hc, Bc, Hc, 0);
      gru_update_rm_k<<<ew_grid(BHe), 256, 0, stream>>>(rpre, mstate, rmb16, BHe);
    }
  }

  // ---- 4. q path ----
  build_q_in_k<<<ew_grid((long)DBr * (Hc + Lc)), 256, 0, stream>>>(mf, tree, qin);
  gemm(qin, Hc + Lc, Hc + Lc, nullptr, 0, 0, wwb, Hc + Lc, nullptr, 0,
       Wb, nullptr, hq, Hc, DBr, Hc, 1);                      // relu, bf16 out
  gemm(hq, Hc, Hc, nullptr, 0, 0, wob, Hc, nullptr, 0,
       Wob, qmat, nullptr, Vc, DBr, Vc, 0);                   // f32 logits
  q_reduce_k<<<DBr, 256, 0, stream>>>(qmat, wid, qlA, qcA);

  // ---- 5. p path (reuses q scratch; stream-ordered after q_reduce) ----
  build_p_in_k<<<ew_grid((long)PRr * (2 * Hc + Lc)), 256, 0, stream>>>(xb, mf, mr, tree, pin);
  gemm(pin, 2 * Hc + Lc, 2 * Hc + Lc, nullptr, 0, 0, uwb, 2 * Hc + Lc, nullptr, 0,
       Ub, nullptr, hp, Hc, PRr, Hc, 1);                      // relu, bf16 out
  p_dot_k<<<PRr, 256, 0, stream>>>(hp, Usw, Usb, plA, pcA);

  // ---- 6. final scalars ----
  final_reduce_k<<<1, 256, 0, stream>>>(qlA, qcA, plA, pcA, out);
}
